// DiffMLAAttention_2680059593234
// MI455X (gfx1250) — matrix operand order repaired and checked by reading, not compiled, not run on hardware
//
#include <hip/hip_runtime.h>

// ---------------------------------------------------------------------------
// CDNA5 (gfx1250) Differential MLA attention forward, bf16 WMMA everywhere.
// Wave32. WMMA shape 16x16x32 bf16 -> f32.  K tiles in attention are loaded
// via the Tensor Data Mover (TENSOR_LOAD_TO_LDS + s_wait_tensorcnt).
// ---------------------------------------------------------------------------

typedef __attribute__((ext_vector_type(16))) __bf16 bf16x16;
typedef __attribute__((ext_vector_type(8)))  float  f32x8;
typedef __attribute__((ext_vector_type(4)))  unsigned int u32x4;
typedef __attribute__((ext_vector_type(8)))  int i32x8;
typedef __attribute__((ext_vector_type(4)))  int i32x4;

union Frag16 { bf16x16 v; u32x4 q[2]; };

#define WMMA_BF16(A, B, C) \
  __builtin_amdgcn_wmma_f32_16x16x32_bf16(false, (A), false, (B), (short)0, (C), false, false)

// ---------------------------------------------------------------------------
// TDM: async DMA of a 2D bf16 tile (rows x 128 cols, row stride in elements)
// from global memory into LDS at byte offset lds_off. Issued by one wave.
// Descriptor packing per CDNA5 ISA sec 8.3/8.4 (D# group0/group1).
// ---------------------------------------------------------------------------
__device__ __forceinline__ void tdm_load_tile_2d(const __bf16* gsrc,
                                                 unsigned lds_off,
                                                 unsigned rows,
                                                 unsigned row_stride_elems) {
  unsigned long long ga = (unsigned long long)(const void*)gsrc;
  unsigned ga_lo = __builtin_amdgcn_readfirstlane((unsigned)(ga & 0xFFFFFFFFu));
  unsigned ga_hi = __builtin_amdgcn_readfirstlane((unsigned)(ga >> 32));
  unsigned loff  = __builtin_amdgcn_readfirstlane(lds_off);

  u32x4 g0;
  g0[0] = 1u;                                   // count=1 (valid descriptor)
  g0[1] = loff;                                 // lds_addr (bytes)
  g0[2] = ga_lo;                                // global_addr[31:0]
  g0[3] = (ga_hi & 0x01FFFFFFu) | (2u << 30);   // global_addr[56:32] | type=2

  i32x8 g1;
  g1[0] = (int)(1u << 16);                      // data_size=1 -> 2 bytes
  g1[1] = (int)(128u << 16);                    // tensor_dim0[15:0]=128 in [31:16]
  g1[2] = (int)(rows << 16);                    // tensor_dim1[15:0] in [31:16]
  g1[3] = (int)(128u << 16);                    // tile_dim0=128 in [31:16]
  g1[4] = (int)rows;                            // tile_dim1 in [15:0], tile_dim2=0
  g1[5] = (int)row_stride_elems;                // tensor_dim0_stride[31:0]
  g1[6] = 0;                                    // stride0[47:32] | stride1[15:0]
  g1[7] = 0;                                    // stride1[47:32]

  i32x4 g2 = {0, 0, 0, 0};
  i32x4 g3 = {0, 0, 0, 0};
#if defined(__clang_major__) && __clang_major__ >= 23
  i32x8 g4 = {0, 0, 0, 0, 0, 0, 0, 0};
  __builtin_amdgcn_tensor_load_to_lds(g0, g1, g2, g3, g4, 0);
#else
  __builtin_amdgcn_tensor_load_to_lds(g0, g1, g2, g3, 0);
#endif
}

// ---------------------------------------------------------------------------
// Elementwise f32 -> bf16
// ---------------------------------------------------------------------------
__global__ void cvt_f32_bf16_kernel(const float* __restrict__ src,
                                    __bf16* __restrict__ dst, int n) {
  int i = blockIdx.x * blockDim.x + threadIdx.x;
  if (i < n) dst[i] = (__bf16)src[i];
}

// ---------------------------------------------------------------------------
// Generic bf16 GEMM: C(f32, MxN) = A(bf16, MxK row-major) @ B(bf16, KxN row-major)
// Optional bf16 copy of C. M,N multiples of 128; K multiple of 32.
// Block tile 128x128x32; 8 waves in 4(M) x 2(N); wave tile 32x64 = 2x4 WMMA tiles.
// ---------------------------------------------------------------------------
__global__ __launch_bounds__(256) void gemm_bf16_wmma_kernel(
    const __bf16* __restrict__ A, const __bf16* __restrict__ B,
    float* __restrict__ C, __bf16* __restrict__ Cb,
    int M, int N, int K) {
  (void)M;
  __shared__ __bf16 ldsA[128][32];   // (m, k)
  __shared__ __bf16 ldsBT[128][32];  // (n, k)  transposed on store

  const int tid  = threadIdx.x;
  const int lane = tid & 31;
  const int wave = tid >> 5;   // 0..7
  const int wm   = wave >> 1;  // 0..3
  const int wn   = wave & 1;   // 0..1
  const int half = lane >> 4;  // 0/1
  const int l15  = lane & 15;

  const int bm = blockIdx.y * 128;
  const int bn = blockIdx.x * 128;

  f32x8 acc[2][4];
#pragma unroll
  for (int i = 0; i < 2; ++i)
#pragma unroll
    for (int j = 0; j < 4; ++j)
#pragma unroll
      for (int e = 0; e < 8; ++e) acc[i][j][e] = 0.0f;

  const int arow = tid >> 1;            // 0..127
  const int acol = (tid & 1) * 16;      // 0 / 16
  const int bk   = tid >> 3;            // 0..31
  const int bnc  = (tid & 7) * 16;      // 0..112

  for (int k0 = 0; k0 < K; k0 += 32) {
    // A tile: each thread copies 16 contiguous bf16 (32 bytes)
    {
      const u32x4* src = (const u32x4*)(A + (size_t)(bm + arow) * K + k0 + acol);
      u32x4 a0 = src[0];
      u32x4 a1 = src[1];
      *(u32x4*)&ldsA[arow][acol]     = a0;
      *(u32x4*)&ldsA[arow][acol + 8] = a1;
      if (k0 + 32 < K)
        __builtin_prefetch(A + (size_t)(bm + arow) * K + k0 + 32 + acol, 0, 1);
    }
    // B tile: read 16 contiguous along N, store transposed (n, k)
    {
      const __bf16* src = B + (size_t)(k0 + bk) * N + bn + bnc;
#pragma unroll
      for (int j = 0; j < 16; ++j) ldsBT[bnc + j][bk] = src[j];
      if (k0 + 32 < K)
        __builtin_prefetch(B + (size_t)(k0 + 32 + bk) * N + bn + bnc, 0, 1);
    }
    __syncthreads();

    Frag16 afrag[2], bfrag[4];
#pragma unroll
    for (int mt = 0; mt < 2; ++mt) {
      int m = wm * 32 + mt * 16 + l15;
      afrag[mt].q[0] = *(const u32x4*)&ldsA[m][16 * half];
      afrag[mt].q[1] = *(const u32x4*)&ldsA[m][16 * half + 8];
    }
#pragma unroll
    for (int nt = 0; nt < 4; ++nt) {
      int n = wn * 64 + nt * 16 + l15;
      bfrag[nt].q[0] = *(const u32x4*)&ldsBT[n][16 * half];
      bfrag[nt].q[1] = *(const u32x4*)&ldsBT[n][16 * half + 8];
    }
#pragma unroll
    for (int mt = 0; mt < 2; ++mt)
#pragma unroll
      for (int nt = 0; nt < 4; ++nt)
        acc[mt][nt] = WMMA_BF16(afrag[mt].v, bfrag[nt].v, acc[mt][nt]);
    __syncthreads();
  }

#pragma unroll
  for (int mt = 0; mt < 2; ++mt)
#pragma unroll
    for (int nt = 0; nt < 4; ++nt)
#pragma unroll
      for (int r = 0; r < 8; ++r) {
        int row = bm + wm * 32 + mt * 16 + r + 8 * half;
        int col = bn + wn * 64 + nt * 16 + l15;
        float v = acc[mt][nt][r];
        C[(size_t)row * N + col] = v;
        if (Cb) Cb[(size_t)row * N + col] = (__bf16)v;
      }
}

// ---------------------------------------------------------------------------
// RoPE: in (L, ncols) f32, heads of 128; out bf16.
// ---------------------------------------------------------------------------
__global__ void rope_bf16_kernel(const float* __restrict__ in,
                                 __bf16* __restrict__ out, int ncols, int total) {
  int idx = blockIdx.x * blockDim.x + threadIdx.x;
  if (idx >= total) return;
  int l = idx / ncols;
  int c = idx - l * ncols;
  int d = c & 127;
  int fi = d & 63;
  float inv = __powf(10000.0f, -(float)fi / 64.0f);
  float ang = (float)l * inv;
  float cs = __cosf(ang), sn = __sinf(ang);
  float t = in[idx];
  float other = (d < 64) ? in[idx + 64] : in[idx - 64];
  float o = (d < 64) ? (t * cs - other * sn) : (t * cs + other * sn);
  out[idx] = (__bf16)o;
}

// ---------------------------------------------------------------------------
// lambda = sigmoid(x @ W_lam + b_lam) ; (L,16)
// ---------------------------------------------------------------------------
__global__ void lambda_kernel(const float* __restrict__ x,
                              const float* __restrict__ Wl,
                              const float* __restrict__ bl,
                              float* __restrict__ lam, int L) {
  int idx = blockIdx.x * blockDim.x + threadIdx.x;
  if (idx >= L * 16) return;
  int l = idx >> 4, h = idx & 15;
  float s = bl[h];
  const float* xr = x + (size_t)l * 2048;
  for (int d = 0; d < 2048; ++d) s += xr[d] * Wl[d * 16 + h];
  lam[idx] = 1.0f / (1.0f + __expf(-s));
}

// ---------------------------------------------------------------------------
// Flash attention, causal, one (q-head, 128-row q-block) per workgroup.
// Qb: (L, 4096) bf16 (post-RoPE, 32 heads of 128)
// Kb: (L, 2048) bf16 (post-RoPE, 16 heads)
// Vb: (L, 2048) bf16
// O : (32, L, 128) f32
// Tiles: Br=128 (16 rows/wave), Bc=64, DH=128.
// K tile is DMA'd into LDS by the Tensor Data Mover; V tile is transposed
// manually (TDM cannot transpose).
// ---------------------------------------------------------------------------
__global__ __launch_bounds__(256) void mla_flash_attn_kernel(
    const __bf16* __restrict__ Qb, const __bf16* __restrict__ Kb,
    const __bf16* __restrict__ Vb, float* __restrict__ O, int L) {
  // Single static LDS block so the K tile sits at LDS byte offset 0.
  __shared__ __align__(128) char smem[3 * 16384];
  __bf16 (*ldsK)[128]    = (__bf16(*)[128])(smem);            // (key, d)   @0
  __bf16 (*ldsVt)[64]    = (__bf16(*)[64])(smem + 16384);     // (d, key)
  __bf16 (*ldsP)[16][64] = (__bf16(*)[16][64])(smem + 32768); // per-wave P

  const int hq  = blockIdx.x;   // 0..31 (query head)
  const int qb  = blockIdx.y;   // 0..15
  const int h   = hq >> 1;      // kv head
  const int tid = threadIdx.x;
  const int lane = tid & 31;
  const int wave = tid >> 5;
  const int half = lane >> 4;
  const int l15  = lane & 15;
  const float scale = 0.08838834764831845f;  // 1/sqrt(128)
  const float NEG_INF = -__builtin_inff();

  // Scalar (SGPR) predicate: first wave issues the TDM ops.
  const bool is_wave0 = (__builtin_amdgcn_readfirstlane(tid) < 32);

  // Preload Q fragments (this wave's 16 rows, all 4 k-steps of DH)
  Frag16 qfrag[4];
  {
    const __bf16* qrow = Qb + (size_t)(qb * 128 + wave * 16 + l15) * 4096 + hq * 128;
#pragma unroll
    for (int ks = 0; ks < 4; ++ks) {
      qfrag[ks].q[0] = *(const u32x4*)(qrow + ks * 32 + 16 * half);
      qfrag[ks].q[1] = *(const u32x4*)(qrow + ks * 32 + 16 * half + 8);
    }
  }

  f32x8 oacc[8];
#pragma unroll
  for (int dt = 0; dt < 8; ++dt)
#pragma unroll
    for (int e = 0; e < 8; ++e) oacc[dt][e] = 0.0f;
  float mrow[8], lrow[8];
#pragma unroll
  for (int r = 0; r < 8; ++r) { mrow[r] = NEG_INF; lrow[r] = 0.0f; }

  const int ldrow = tid >> 2;         // 0..63 (key row within tile)
  const int ldcc  = (tid & 3) * 32;   // 0/32/64/96 (d chunk)

  const int kbmax = 2 * qb + 1;
  for (int kb = 0; kb <= kbmax; ++kb) {
    // Async K tile (64 x 128) via Tensor Data Mover into LDS offset 0.
    if (is_wave0)
      tdm_load_tile_2d(Kb + (size_t)(kb * 64) * 2048 + h * 128, 0u, 64u, 2048u);

    // V tile transposed into (d, key) by all threads (overlaps the DMA).
    {
      const __bf16* vsrc = Vb + (size_t)(kb * 64 + ldrow) * 2048 + h * 128 + ldcc;
#pragma unroll
      for (int j = 0; j < 32; ++j) ldsVt[ldcc + j][ldrow] = vsrc[j];
    }
    if (is_wave0) __builtin_amdgcn_s_wait_tensorcnt(0);
    __syncthreads();

    // S = Q @ K^T  (16 rows x 64 keys per wave)
    f32x8 sacc[4];
#pragma unroll
    for (int nt = 0; nt < 4; ++nt)
#pragma unroll
      for (int e = 0; e < 8; ++e) sacc[nt][e] = 0.0f;
#pragma unroll
    for (int ks = 0; ks < 4; ++ks) {
#pragma unroll
      for (int nt = 0; nt < 4; ++nt) {
        Frag16 bf;
        int n = nt * 16 + l15;
        bf.q[0] = *(const u32x4*)&ldsK[n][ks * 32 + 16 * half];
        bf.q[1] = *(const u32x4*)&ldsK[n][ks * 32 + 16 * half + 8];
        sacc[nt] = WMMA_BF16(qfrag[ks].v, bf.v, sacc[nt]);
      }
    }

    // Scale + causal mask
    const int mbase = qb * 128 + wave * 16 + 8 * half;
#pragma unroll
    for (int nt = 0; nt < 4; ++nt) {
      int nglob = kb * 64 + nt * 16 + l15;
#pragma unroll
      for (int r = 0; r < 8; ++r) {
        float s = sacc[nt][r] * scale;
        if (nglob > mbase + r) s = NEG_INF;
        sacc[nt][r] = s;
      }
    }

    // Online softmax (row stats replicated across 16-lane halves)
    float alpha[8];
#pragma unroll
    for (int r = 0; r < 8; ++r) {
      float mx = sacc[0][r];
#pragma unroll
      for (int nt = 1; nt < 4; ++nt) mx = fmaxf(mx, sacc[nt][r]);
#pragma unroll
      for (int off = 1; off < 16; off <<= 1)
        mx = fmaxf(mx, __shfl_xor(mx, off, 32));
      float mn = fmaxf(mrow[r], mx);
      alpha[r] = __expf(mrow[r] - mn);
      mrow[r] = mn;
    }
    float rsum[8];
#pragma unroll
    for (int r = 0; r < 8; ++r) rsum[r] = 0.0f;
#pragma unroll
    for (int nt = 0; nt < 4; ++nt)
#pragma unroll
      for (int r = 0; r < 8; ++r) {
        float p = __expf(sacc[nt][r] - mrow[r]);
        sacc[nt][r] = p;
        rsum[r] += p;
      }
#pragma unroll
    for (int r = 0; r < 8; ++r) {
      float sm = rsum[r];
#pragma unroll
      for (int off = 1; off < 16; off <<= 1) sm += __shfl_xor(sm, off, 32);
      lrow[r] = lrow[r] * alpha[r] + sm;
    }
#pragma unroll
    for (int dt = 0; dt < 8; ++dt)
#pragma unroll
      for (int r = 0; r < 8; ++r) oacc[dt][r] *= alpha[r];

    // P -> LDS (per-wave region, no cross-wave barrier needed)
#pragma unroll
    for (int nt = 0; nt < 4; ++nt)
#pragma unroll
      for (int r = 0; r < 8; ++r)
        ldsP[wave][r + 8 * half][nt * 16 + l15] = (__bf16)sacc[nt][r];

    // O += P @ V   (k-dim = 64 keys -> 2 k-steps)
#pragma unroll
    for (int ks = 0; ks < 2; ++ks) {
      Frag16 pa;
      pa.q[0] = *(const u32x4*)&ldsP[wave][l15][ks * 32 + 16 * half];
      pa.q[1] = *(const u32x4*)&ldsP[wave][l15][ks * 32 + 16 * half + 8];
#pragma unroll
      for (int dt = 0; dt < 8; ++dt) {
        Frag16 vb;
        int d = dt * 16 + l15;
        vb.q[0] = *(const u32x4*)&ldsVt[d][ks * 32 + 16 * half];
        vb.q[1] = *(const u32x4*)&ldsVt[d][ks * 32 + 16 * half + 8];
        oacc[dt] = WMMA_BF16(pa.v, vb.v, oacc[dt]);
      }
    }
    __syncthreads();
  }

  // Normalize and store O
#pragma unroll
  for (int dt = 0; dt < 8; ++dt)
#pragma unroll
    for (int r = 0; r < 8; ++r) {
      int row = qb * 128 + wave * 16 + r + 8 * half;
      float v = oacc[dt][r] / lrow[r];
      O[((size_t)hq * L + row) * 128 + dt * 16 + l15] = v;
    }
}

// ---------------------------------------------------------------------------
// merged(l, h*128+d) = attn[2h](l,d) - lam(l,h) * attn[2h+1](l,d)   -> bf16
// ---------------------------------------------------------------------------
__global__ void merge_kernel(const float* __restrict__ attn,
                             const float* __restrict__ lam,
                             __bf16* __restrict__ merged, int L) {
  int idx = blockIdx.x * blockDim.x + threadIdx.x;
  if (idx >= L * 2048) return;
  int l = idx >> 11;
  int c = idx & 2047;
  int h = c >> 7;
  int d = c & 127;
  float a1 = attn[((size_t)(2 * h) * L + l) * 128 + d];
  float a2 = attn[((size_t)(2 * h + 1) * L + l) * 128 + d];
  merged[idx] = (__bf16)(a1 - lam[l * 16 + h] * a2);
}

// ---------------------------------------------------------------------------
// Host launcher
// ---------------------------------------------------------------------------
extern "C" void kernel_launch(void* const* d_in, const int* in_sizes, int n_in,
                              void* d_out, int out_size, void* d_ws, size_t ws_size,
                              hipStream_t stream) {
  (void)in_sizes; (void)n_in; (void)out_size; (void)ws_size;
  const int L = 2048, D = 2048, DC = 512, DCQ = 1024, DQ2 = 4096;

  const float* x    = (const float*)d_in[0];
  const float* wdkv = (const float*)d_in[1];
  const float* wuk  = (const float*)d_in[2];
  const float* wuv  = (const float*)d_in[3];
  const float* wdq  = (const float*)d_in[4];
  const float* wuq  = (const float*)d_in[5];
  const float* wlam = (const float*)d_in[6];
  const float* blam = (const float*)d_in[7];
  const float* wout = (const float*)d_in[8];
  float* out = (float*)d_out;

  char* ws = (char*)d_ws;
  size_t off = 0;
  auto alloc = [&](size_t bytes) -> void* {
    void* p = ws + off;
    off = (off + bytes + 255) & ~(size_t)255;
    return p;
  };

  __bf16* xb     = (__bf16*)alloc((size_t)L * D * 2);
  __bf16* wdkv_b = (__bf16*)alloc((size_t)D * DC * 2);
  __bf16* wuk_b  = (__bf16*)alloc((size_t)DC * D * 2);
  __bf16* wuv_b  = (__bf16*)alloc((size_t)DC * D * 2);
  __bf16* wdq_b  = (__bf16*)alloc((size_t)D * DCQ * 2);
  __bf16* wuq_b  = (__bf16*)alloc((size_t)DCQ * DQ2 * 2);
  __bf16* wout_b = (__bf16*)alloc((size_t)D * D * 2);
  float*  ckv_f  = (float*) alloc((size_t)L * DC * 4);
  __bf16* ckv_b  = (__bf16*)alloc((size_t)L * DC * 2);
  float*  cq_f   = (float*) alloc((size_t)L * DCQ * 4);
  __bf16* cq_b   = (__bf16*)alloc((size_t)L * DCQ * 2);
  float*  kf     = (float*) alloc((size_t)L * D * 4);
  float*  vf     = (float*) alloc((size_t)L * D * 4);
  float*  qf     = (float*) alloc((size_t)L * DQ2 * 4);
  __bf16* k_b    = (__bf16*)alloc((size_t)L * D * 2);
  __bf16* v_b    = (__bf16*)alloc((size_t)L * D * 2);
  __bf16* q_b    = (__bf16*)alloc((size_t)L * DQ2 * 2);
  float*  lam    = (float*) alloc((size_t)L * 16 * 4);
  float*  attn   = (float*) alloc((size_t)32 * L * 128 * 4);
  __bf16* merged = (__bf16*)alloc((size_t)L * D * 2);

  auto cvt = [&](const float* s, __bf16* d, int n) {
    cvt_f32_bf16_kernel<<<dim3((n + 255) / 256), 256, 0, stream>>>(s, d, n);
  };

  // 1) Convert inputs/weights to bf16
  cvt(x,    xb,     L * D);
  cvt(wdkv, wdkv_b, D * DC);
  cvt(wuk,  wuk_b,  DC * D);
  cvt(wuv,  wuv_b,  DC * D);
  cvt(wdq,  wdq_b,  D * DCQ);
  cvt(wuq,  wuq_b,  DCQ * DQ2);
  cvt(wout, wout_b, D * D);

  // 2) Projection GEMMs
  gemm_bf16_wmma_kernel<<<dim3(DC / 128, L / 128), 256, 0, stream>>>(
      xb, wdkv_b, ckv_f, ckv_b, L, DC, D);                       // c_kv
  gemm_bf16_wmma_kernel<<<dim3(D / 128, L / 128), 256, 0, stream>>>(
      ckv_b, wuk_b, kf, nullptr, L, D, DC);                      // k (pre-RoPE)
  gemm_bf16_wmma_kernel<<<dim3(D / 128, L / 128), 256, 0, stream>>>(
      ckv_b, wuv_b, vf, v_b, L, D, DC);                          // v
  gemm_bf16_wmma_kernel<<<dim3(DCQ / 128, L / 128), 256, 0, stream>>>(
      xb, wdq_b, cq_f, cq_b, L, DCQ, D);                         // c_q
  gemm_bf16_wmma_kernel<<<dim3(DQ2 / 128, L / 128), 256, 0, stream>>>(
      cq_b, wuq_b, qf, nullptr, L, DQ2, DCQ);                    // q (pre-RoPE)

  // 3) RoPE + lambda
  rope_bf16_kernel<<<dim3((L * DQ2) / 256), 256, 0, stream>>>(qf, q_b, DQ2, L * DQ2);
  rope_bf16_kernel<<<dim3((L * D) / 256), 256, 0, stream>>>(kf, k_b, D, L * D);
  lambda_kernel<<<dim3((L * 16) / 256), 256, 0, stream>>>(x, wlam, blam, lam, L);

  // 4) Flash attention: 32 query heads x 16 q-blocks
  mla_flash_attn_kernel<<<dim3(32, L / 128), 256, 0, stream>>>(q_b, k_b, v_b, attn, L);

  // 5) Differential merge + output projection
  merge_kernel<<<dim3((L * D) / 256), 256, 0, stream>>>(attn, lam, merged, L);
  gemm_bf16_wmma_kernel<<<dim3(D / 128, L / 128), 256, 0, stream>>>(
      merged, wout_b, out, nullptr, L, D, D);
}
